// EncoderLayer_39161511805522
// MI455X (gfx1250) — matrix operand order repaired and checked by reading, not compiled, not run on hardware
//
#include <hip/hip_runtime.h>
#include <hip/hip_bf16.h>
#include <math.h>

typedef __attribute__((ext_vector_type(16))) __bf16 v16bf;
typedef __attribute__((ext_vector_type(8)))  __bf16 v8bf;
typedef __attribute__((ext_vector_type(8)))  float  v8f;

#define HID   512
#define HEADS 8
#define HD    64
#define PF    2048
#define LSEQ  512
#define BATCH 2
#define EPS   1e-5f

#if __has_builtin(__builtin_amdgcn_tanhf)
#define TANHF(x) __builtin_amdgcn_tanhf(x)
#else
#define TANHF(x) tanhf(x)
#endif

// ---------------------------------------------------------------------------
// Generic GEMM: C(MxN) = A(MxK) @ W(KxN) + bias. N, K, RELU are compile-time.
// Block = 128 threads (4 waves). Tile 64x64, K-step 32, bf16 WMMA, f32 acc.
// Double-buffered LDS pipeline: global loads for tile i+1 issue before the
// WMMA chain of tile i; one barrier per K-step.
// ---------------------------------------------------------------------------
#define BM 64
#define BN 64
#define BK 32
#define AST 40   // LDS row stride (mult of 8 -> 16B-aligned fragment halves)
#define BST 40

union PK4 { __bf16 h[4]; unsigned long long u; };

template <int RELU, int N, int K>
__global__ __launch_bounds__(128)
void gemm_bf16_wmma(const float* __restrict__ A, const float* __restrict__ W,
                    const float* __restrict__ bias, float* __restrict__ C)
{
    __shared__ __attribute__((aligned(16))) __bf16 As [2][BM * AST]; // [m][k]
    __shared__ __attribute__((aligned(16))) __bf16 Bst[2][BN * BST]; // [n][k]

    const int tid  = threadIdx.x;
    const int wave = tid >> 5;
    const int lane = tid & 31;
    const int ln   = lane & 15;   // fragment row/col index
    const int lh   = lane >> 4;   // lane half

    const int rowBase = blockIdx.y * BM;
    const int colBase = blockIdx.x * BN;

    // Per-thread staging coordinates (loop-invariant)
    int ar[4], ac[4], bc[4], bk[4];
    #pragma unroll
    for (int j = 0; j < 4; ++j) {
        int q = j * 128 + tid;
        ar[j] = q >> 3;  ac[j] = (q & 7) * 4;   // A: 64x32, quads along k
        bc[j] = q & 63;  bk[j] = (q >> 6) * 4;  // B: quads along k, cols along n
    }

    float4 fa[4];
    float  wb[4][4];

    auto load_tile = [&](int k0) {
        #pragma unroll
        for (int j = 0; j < 4; ++j)
            fa[j] = *(const float4*)&A[(rowBase + ar[j]) * K + k0 + ac[j]];
        #pragma unroll
        for (int j = 0; j < 4; ++j)
            #pragma unroll
            for (int i = 0; i < 4; ++i)
                wb[j][i] = W[(k0 + bk[j] + i) * N + colBase + bc[j]];
    };
    auto store_tile = [&](int buf) {
        #pragma unroll
        for (int j = 0; j < 4; ++j) {
            PK4 pk;
            pk.h[0] = (__bf16)fa[j].x; pk.h[1] = (__bf16)fa[j].y;
            pk.h[2] = (__bf16)fa[j].z; pk.h[3] = (__bf16)fa[j].w;
            *(unsigned long long*)&As[buf][ar[j] * AST + ac[j]] = pk.u;
        }
        #pragma unroll
        for (int j = 0; j < 4; ++j) {
            PK4 pk;
            #pragma unroll
            for (int i = 0; i < 4; ++i) pk.h[i] = (__bf16)wb[j][i];
            *(unsigned long long*)&Bst[buf][bc[j] * BST + bk[j]] = pk.u;
        }
    };

    v8f acc[4] = {};

    // Prologue: stage tile 0 into buffer 0
    load_tile(0);
    store_tile(0);
    __syncthreads();

    constexpr int NSTEP = K / BK;
    for (int s = 0; s < NSTEP; ++s) {
        const int cur = s & 1;
        const bool hasNext = (s + 1) < NSTEP;

        // Issue next tile's global loads now; they fly under the WMMA chain.
        if (hasNext) load_tile((s + 1) * BK);
        // Prefetch tile s+2 toward the WGP (global_prefetch_b8).
        if (s + 2 < NSTEP) {
            int k2 = (s + 2) * BK;
            __builtin_prefetch(&A[(rowBase + (tid & 63)) * K + k2 + (tid >> 6) * 16], 0, 3);
            __builtin_prefetch(&W[(k2 + (tid & 31)) * N + colBase + (tid >> 5) * 16], 0, 3);
        }

        // Fragments from current buffer.
        union { v16bf v; v8bf h[2]; } ua;
        {
            const __bf16* ap = &As[cur][(wave * 16 + ln) * AST];
            ua.h[0] = *(const v8bf*)&ap[lh * 8];
            ua.h[1] = *(const v8bf*)&ap[16 + lh * 8];
        }
        union { v16bf v; v8bf h[2]; } ubf[4];
        #pragma unroll
        for (int t = 0; t < 4; ++t) {
            const __bf16* bp = &Bst[cur][(t * 16 + ln) * BST + lh * 8];
            ubf[t].h[0] = *(const v8bf*)&bp[0];
            ubf[t].h[1] = *(const v8bf*)&bp[16];
        }
        #pragma unroll
        for (int t = 0; t < 4; ++t)
            acc[t] = __builtin_amdgcn_wmma_f32_16x16x32_bf16(
                false, ua.v, false, ubf[t].v, (short)0, acc[t], false, false);

        // Convert + store next tile into the other buffer, then one barrier.
        if (hasNext) store_tile(cur ^ 1);
        __syncthreads();
    }

    // Epilogue: C layout VGPR r -> row r + 8*lh, lanes -> col ln
    #pragma unroll
    for (int t = 0; t < 4; ++t) {
        int n = colBase + t * 16 + ln;
        const float bv = bias[n];
        #pragma unroll
        for (int r = 0; r < 8; ++r) {
            int m = rowBase + wave * 16 + r + 8 * lh;
            float v = acc[t][r] + bv;
            if (RELU) v = v > 0.f ? v : 0.f;
            C[m * N + n] = v;
        }
    }
}

// ---------------------------------------------------------------------------
// Fused additive attention: one block per (b,h,q).
// energy[k] = sum_d Vw[d]*tanh(Qa[q,d]+Ka[k,d]) + Vb ; mask ; softmax ; P@V
// ---------------------------------------------------------------------------
__global__ __launch_bounds__(256)
void additive_attention(const float* __restrict__ Qa, const float* __restrict__ Ka,
                        const float* __restrict__ V,  const float* __restrict__ Vw,
                        const float* __restrict__ Vb, const int* __restrict__ mask,
                        float* __restrict__ X)
{
    __shared__ float qrow[HD];
    __shared__ float vw[HD];
    __shared__ float sc[LSEQ];
    __shared__ float red[256];
    __shared__ float pv[4][HD];

    const int q = blockIdx.x, h = blockIdx.y, b = blockIdx.z;
    const int t = threadIdx.x;

    if (t < HD) {
        qrow[t] = Qa[(((b * LSEQ + q) * HEADS) + h) * HD + t];
        vw[t]   = Vw[t];
    }
    __syncthreads();

    const float vb = Vb[0];
    for (int kk = t; kk < LSEQ; kk += 256) {
        const float* ka = &Ka[(((b * LSEQ + kk) * HEADS) + h) * HD];
        float s = vb;
        #pragma unroll 8
        for (int d = 0; d < HD; ++d)
            s += vw[d] * TANHF(qrow[d] + ka[d]);
        if (mask[b * LSEQ + kk] == 0) s = -1e10f;
        sc[kk] = s;
    }
    __syncthreads();

    // row max
    red[t] = fmaxf(sc[t], sc[t + 256]);
    __syncthreads();
    for (int s = 128; s > 0; s >>= 1) {
        if (t < s) red[t] = fmaxf(red[t], red[t + s]);
        __syncthreads();
    }
    const float m = red[0];
    __syncthreads();

    // exp + sum
    float ls = 0.f;
    for (int kk = t; kk < LSEQ; kk += 256) {
        float e = __expf(sc[kk] - m);
        sc[kk] = e;
        ls += e;
    }
    red[t] = ls;
    __syncthreads();
    for (int s = 128; s > 0; s >>= 1) {
        if (t < s) red[t] += red[t + s];
        __syncthreads();
    }
    const float inv = 1.f / red[0];
    __syncthreads();

    // P @ V : 64 d-columns x 4 k-slices
    const int d = t & 63, slice = t >> 6;
    float acc = 0.f;
    const int kbeg = slice * 128;
    #pragma unroll 4
    for (int kk = kbeg; kk < kbeg + 128; ++kk)
        acc += sc[kk] * V[(b * LSEQ + kk) * HID + h * HD + d];
    pv[slice][d] = acc;
    __syncthreads();
    if (t < HD)
        X[(b * LSEQ + q) * HID + h * HD + t] =
            (pv[0][t] + pv[1][t] + pv[2][t] + pv[3][t]) * inv;
}

// ---------------------------------------------------------------------------
// out = LayerNorm(X + R) * g + b   (one block per 512-wide row)
// ---------------------------------------------------------------------------
__global__ __launch_bounds__(256)
void add_layernorm(const float* __restrict__ X, const float* __restrict__ R,
                   const float* __restrict__ g, const float* __restrict__ bb,
                   float* __restrict__ O)
{
    __shared__ float red[256];
    const int row = blockIdx.x, t = threadIdx.x;
    const float x0 = X[row * HID + t]       + R[row * HID + t];
    const float x1 = X[row * HID + t + 256] + R[row * HID + t + 256];
    red[t] = x0 + x1;
    __syncthreads();
    for (int s = 128; s > 0; s >>= 1) { if (t < s) red[t] += red[t + s]; __syncthreads(); }
    const float mean = red[0] * (1.f / HID);
    __syncthreads();
    const float d0 = x0 - mean, d1 = x1 - mean;
    red[t] = d0 * d0 + d1 * d1;
    __syncthreads();
    for (int s = 128; s > 0; s >>= 1) { if (t < s) red[t] += red[t + s]; __syncthreads(); }
    const float rstd = rsqrtf(red[0] * (1.f / HID) + EPS);
    O[row * HID + t]       = d0 * rstd * g[t]       + bb[t];
    O[row * HID + t + 256] = d1 * rstd * g[t + 256] + bb[t + 256];
}

// ---------------------------------------------------------------------------
extern "C" void kernel_launch(void* const* d_in, const int* in_sizes, int n_in,
                              void* d_out, int out_size, void* d_ws, size_t ws_size,
                              hipStream_t stream) {
    const float* src   = (const float*)d_in[0];
    const int*   smask = (const int*)  d_in[1];
    const float* Wq = (const float*)d_in[2],  *bq = (const float*)d_in[3];
    const float* Wk = (const float*)d_in[4],  *bk = (const float*)d_in[5];
    const float* Wv = (const float*)d_in[6],  *bv = (const float*)d_in[7];
    const float* Wo = (const float*)d_in[8],  *bo = (const float*)d_in[9];
    const float* Wa = (const float*)d_in[10], *ba = (const float*)d_in[11];
    const float* Ua = (const float*)d_in[12], *ub = (const float*)d_in[13];
    const float* Vw = (const float*)d_in[14], *Vb = (const float*)d_in[15];
    const float* g1 = (const float*)d_in[16], *b1n = (const float*)d_in[17];
    const float* g2 = (const float*)d_in[18], *b2n = (const float*)d_in[19];
    const float* W1 = (const float*)d_in[20], *b1 = (const float*)d_in[21];
    const float* W2 = (const float*)d_in[22], *b2 = (const float*)d_in[23];
    float* out = (float*)d_out;

    const size_t NT = (size_t)BATCH * LSEQ * HID; // 524288 floats
    float* ws = (float*)d_ws;
    float* Q    = ws + 0 * NT;
    float* Kp   = ws + 1 * NT;
    float* Vp   = ws + 2 * NT;
    float* Qa   = ws + 3 * NT;
    float* Ka   = ws + 4 * NT;
    float* X    = ws + 5 * NT;
    float* AO   = ws + 0 * NT;   // reuse Q
    float* src1 = ws + 1 * NT;   // reuse K
    float* F1   = ws + 3 * NT;   // reuse Qa..X (+1 NT): 4*NT floats
    float* F2   = ws + 2 * NT;   // reuse V
    (void)in_sizes; (void)n_in; (void)out_size; (void)ws_size;

    const int M = BATCH * LSEQ;          // 1024
    dim3 blk(128);

    // QKV projections (stored as (B,L,H,HD) == (B,L,HID))
    gemm_bf16_wmma<0, HID, HID><<<dim3(HID/BN, M/BM), blk, 0, stream>>>(src, Wq, bq, Q);
    gemm_bf16_wmma<0, HID, HID><<<dim3(HID/BN, M/BM), blk, 0, stream>>>(src, Wk, bk, Kp);
    gemm_bf16_wmma<0, HID, HID><<<dim3(HID/BN, M/BM), blk, 0, stream>>>(src, Wv, bv, Vp);

    // Per-head Qa/Ka: view (B,L,H,HD) as (B*L*H, 64) rows
    const int MH = M * HEADS;            // 8192
    gemm_bf16_wmma<0, HD, HD><<<dim3(HD/BN, MH/BM), blk, 0, stream>>>(Q,  Wa, ba, Qa);
    gemm_bf16_wmma<0, HD, HD><<<dim3(HD/BN, MH/BM), blk, 0, stream>>>(Kp, Ua, ub, Ka);

    // Fused additive attention -> X in (B,L,HID)
    additive_attention<<<dim3(LSEQ, HEADS, BATCH), dim3(256), 0, stream>>>(
        Qa, Ka, Vp, Vw, Vb, smask, X);

    // Output projection, residual + LN1
    gemm_bf16_wmma<0, HID, HID><<<dim3(HID/BN, M/BM), blk, 0, stream>>>(X, Wo, bo, AO);
    add_layernorm<<<dim3(M), dim3(256), 0, stream>>>(AO, src, g1, b1n, src1);

    // FFN
    gemm_bf16_wmma<1, PF, HID><<<dim3(PF/BN,  M/BM), blk, 0, stream>>>(src1, W1, b1, F1);
    gemm_bf16_wmma<0, HID, PF><<<dim3(HID/BN, M/BM), blk, 0, stream>>>(F1,   W2, b2, F2);
    add_layernorm<<<dim3(M), dim3(256), 0, stream>>>(F2, src1, g2, b2n, out);
}
